// ImprovedxLSTMPredictor_11158325035134
// MI455X (gfx1250) — matrix operand order repaired and checked by reading, not compiled, not run on hardware
//
#include <hip/hip_runtime.h>
#include <math.h>

// ============================================================================
// Persistent xLSTM kernel for MI455X (gfx1250, wave32, WMMA).
//  - 32 blocks x 256 threads (8 wave32s); each block owns a 16-row batch tile.
//  - All weights converted to f16 and pre-swizzled into WMMA B-fragment order
//    in LDS once; the T=1024 recurrence then runs entirely out of LDS/VGPRs.
//  - Every GEMM is v_wmma_f32_16x16x32_f16; activations use native v_tanh_f32
//    and v_rcp_f32 (latency-bound recurrence -> cheap TRANS ops matter).
// ============================================================================

typedef __attribute__((ext_vector_type(16))) _Float16 v16h;
typedef __attribute__((ext_vector_type(8)))  _Float16 v8h;
typedef __attribute__((ext_vector_type(8)))  float    v8f;

#define DEVINL __device__ __forceinline__

static constexpr int Tlen = 1024;

// ---------------- LDS layout ----------------
// f16 region (units: halves). Weight matrices stored as WMMA B fragments:
// fragment (ni,ki) at offset (ni*KT + ki)*512; within a fragment lane l holds
// 16 consecutive halves (one column, K = 8*(l/16)+{0..7} and 16+8*(l/16)+{0..7}).
constexpr int HW_M1_Q = 0;                  // 128x64
constexpr int HW_M1_K = HW_M1_Q + 8192;     // 128x16
constexpr int HW_M1_V = HW_M1_K + 2048;     // 128x16
constexpr int HW_M1_I = HW_M1_V + 2048;     // 192x16
constexpr int HW_M1_F = HW_M1_I + 3072;     // 192x16
constexpr int HW_M1_O = HW_M1_F + 3072;     // 128x64 (row 128 handled separately)
constexpr int HW_M2_Q = HW_M1_O + 8192;     // 64x64
constexpr int HW_M2_K = HW_M2_Q + 4096;     // 64x16
constexpr int HW_M2_V = HW_M2_K + 1024;     // 64x16
constexpr int HW_M2_I = HW_M2_V + 1024;     // 128x16
constexpr int HW_M2_F = HW_M2_I + 2048;     // 128x16
constexpr int HW_M2_O = HW_M2_F + 2048;     // 64x64 (row 64 separate)
constexpr int HW_S1   = HW_M2_O + 4096;     // 4 x 128x64 (i,f,o,c)
constexpr int HW_S2   = HW_S1 + 32768;      // 4 x 128x64
constexpr int HX_N1   = HW_S2 + 32768;      // xn1/comb1  [16][192] f16
constexpr int HX_N2   = HX_N1 + 3072;       // comb2      [16][128]
constexpr int HX_N3   = HX_N2 + 2048;       // comb3      [16][128]
constexpr int HX_N4   = HX_N3 + 2048;       // comb4      [16][128]
constexpr int H_TOTAL = HX_N4 + 2048;       // = 115712 halves

// f32 region (units: floats), base = smem + H_TOTAL*2
constexpr int F_B_M1_Q = 0;        // 64
constexpr int F_B_M1_K = 64;       // 16
constexpr int F_B_M1_V = 80;       // 16
constexpr int F_B_M1_I = 96;       // 16
constexpr int F_B_M1_F = 112;      // 16
constexpr int F_B_M1_O = 128;      // 64
constexpr int F_WO1L   = 192;      // 64 (m_mean column of m1.Wo)
constexpr int F_LNM1_G = 256;      // 64
constexpr int F_LNM1_B = 320;      // 64
constexpr int F_B_M2_Q = 384;
constexpr int F_B_M2_K = 448;
constexpr int F_B_M2_V = 464;
constexpr int F_B_M2_I = 480;
constexpr int F_B_M2_F = 496;
constexpr int F_B_M2_O = 512;
constexpr int F_WO2L   = 576;
constexpr int F_LNM2_G = 640;
constexpr int F_LNM2_B = 704;
constexpr int F_BS1    = 768;      // 4x64 (i,f,o,c)
constexpr int F_LNS1_G = 1024;
constexpr int F_LNS1_B = 1088;
constexpr int F_BS2    = 1152;     // 4x64
constexpr int F_LNS2_G = 1408;
constexpr int F_LNS2_B = 1472;
constexpr int F_LN1_G  = 1536;
constexpr int F_LN1_B  = 1600;
constexpr int F_LN2_G  = 1664;
constexpr int F_LN2_B  = 1728;
constexpr int F_LN3_G  = 1792;
constexpr int F_LN3_B  = 1856;
constexpr int F_WIP    = 1920;     // 128
constexpr int F_BIP    = 2048;     // 128
constexpr int F_QT     = 2176;     // [16][64] (qt / LN scratch)
constexpr int F_OT     = 3200;     // [16][64]
constexpr int F_KT     = 4224;     // [16][16]
constexpr int F_VT     = 4480;     // [16][16]
constexpr int F_IT     = 4736;     // [16][16]
constexpr int F_FT     = 4992;     // [16][16]
constexpr int F_ITG    = 5248;     // [16][64] sLSTM gates
constexpr int F_FTG    = 6272;
constexpr int F_OTG    = 7296;
constexpr int F_CHG    = 8320;
constexpr int F_H1     = 9344;     // [16][64] recurrent states
constexpr int F_H2     = 10368;
constexpr int F_H3     = 11392;
constexpr int F_H4     = 12416;
constexpr int F_C3     = 13440;
constexpr int F_C4     = 14464;
constexpr int F_RED    = 15488;    // 512 reduction scratch
constexpr int F_MM     = 16000;    // 16  (m_mean per row)
constexpr int F_TOTAL  = 16016;

constexpr size_t SMEM_BYTES = (size_t)H_TOTAL * 2 + (size_t)F_TOTAL * 4;

// ---------------- fast activation helpers ----------------
// gfx1250 has native V_TANH_F32 (TRANS op); use it when clang exposes it.
#if __has_builtin(__builtin_amdgcn_tanhf)
DEVINL float fast_tanh(float x) { return __builtin_amdgcn_tanhf(x); }
#else
DEVINL float fast_tanh(float x) { return tanhf(x); }
#endif

// sigmoid via v_exp_f32 + v_rcp_f32 (avoid full IEEE-div expansion)
DEVINL float sig(float x) {
  return __builtin_amdgcn_rcpf(1.0f + __expf(-x));
}

DEVINL v8f vzero() {
  v8f v;
#pragma unroll
  for (int i = 0; i < 8; ++i) v[i] = 0.0f;
  return v;
}

// A fragment: row-major f16 [16][stride], CDNA5 16-bit A layout:
// lane<16 : K = k0+{0..7}, k0+{16..23} ; lane>=16 : K = k0+{8..15}, k0+{24..31}
DEVINL v16h load_a(const _Float16* A, int stride, int k0) {
  int lane = threadIdx.x & 31;
  int row  = lane & 15;
  int hi   = lane >> 4;
  const _Float16* p = A + row * stride + k0 + hi * 8;
  v8h lo = *(const v8h*)p;
  v8h hh = *(const v8h*)(p + 16);
  return __builtin_shufflevector(lo, hh, 0,1,2,3,4,5,6,7,8,9,10,11,12,13,14,15);
}

// B fragment: pre-swizzled, lane reads 16 consecutive halves (32B)
DEVINL v16h load_b(const _Float16* frag) {
  int lane = threadIdx.x & 31;
  const _Float16* p = frag + lane * 16;
  v8h lo = *(const v8h*)p;
  v8h hh = *(const v8h*)(p + 8);
  return __builtin_shufflevector(lo, hh, 0,1,2,3,4,5,6,7,8,9,10,11,12,13,14,15);
}

// Preload ALL fragments, then drain the (acc-serial) WMMA chain; lets every
// ds_load_b128 be outstanding while the matrix pipe works.
template <int KT>
DEVINL v8f gemm(const _Float16* A, int strideA, const _Float16* Bf, v8f acc) {
  v16h a[KT], b[KT];
#pragma unroll
  for (int ki = 0; ki < KT; ++ki) {
    a[ki] = load_a(A, strideA, ki * 32);
    b[ki] = load_b(Bf + ki * 512);
  }
#pragma unroll
  for (int ki = 0; ki < KT; ++ki)
    acc = __builtin_amdgcn_wmma_f32_16x16x32_f16(false, a[ki], false, b[ki],
                                                 (short)0, acc, false, false);
  return acc;
}

// D tile (16x16 f32): VGPR r, lane l -> row = r + 8*(l>>4), col = n0 + (l&15)
template <typename F>
DEVINL void store_tile(float* D, int ldD, int n0, const v8f& acc, F f) {
  int lane = threadIdx.x & 31;
  int col  = n0 + (lane & 15);
  int hi   = lane >> 4;
#pragma unroll
  for (int r = 0; r < 8; ++r) {
    int row = r + 8 * hi;
    D[row * ldD + col] = f(acc[r], row, col);
  }
}

// Swizzle a global f32 [K][N] row-major matrix into LDS f16 B-fragments.
__device__ void load_weight(const float* W, _Float16* dst, int K, int N) {
  int KT = K >> 5;
  int total = K * N;
  for (int e = threadIdx.x; e < total; e += 256) {
    int frag = e >> 9;
    int idx  = e & 511;
    int lane = idx >> 4;
    int h    = idx & 15;
    int ni   = frag / KT;
    int ki   = frag - ni * KT;
    int col  = ni * 16 + (lane & 15);
    int k    = ki * 32 + ((lane >> 4) << 3) + (h & 7) + ((h >> 3) << 4);
    dst[e] = (_Float16)W[k * N + col];
  }
}

__device__ void copyf(const float* src, float* dst, int n) {
  for (int i = threadIdx.x; i < n; i += 256) dst[i] = src[i];
}

// LayerNorm over 64 features, 16 rows, 256 threads. Safe in-place (src==dst).
__device__ void layernorm_16x64(const float* src, float* dst,
                                const float* g, const float* b, float* red) {
  int t = threadIdx.x;
  int row = t >> 4, seg = t & 15;
  const float* s = src + row * 64 + seg * 4;
  float v0 = s[0], v1 = s[1], v2 = s[2], v3 = s[3];
  red[row * 16 + seg]       = v0 + v1 + v2 + v3;
  red[256 + row * 16 + seg] = v0 * v0 + v1 * v1 + v2 * v2 + v3 * v3;
  __syncthreads();
  float sum = 0.f, sq = 0.f;
#pragma unroll
  for (int i = 0; i < 16; ++i) {
    sum += red[row * 16 + i];
    sq  += red[256 + row * 16 + i];
  }
  float mu  = sum * (1.0f / 64.0f);
  float var = sq * (1.0f / 64.0f) - mu * mu;
  float inv = rsqrtf(var + 1e-5f);
  float* d = dst + row * 64 + seg * 4;
  d[0] = g[seg * 4 + 0] * (v0 - mu) * inv + b[seg * 4 + 0];
  d[1] = g[seg * 4 + 1] * (v1 - mu) * inv + b[seg * 4 + 1];
  d[2] = g[seg * 4 + 2] * (v2 - mu) * inv + b[seg * 4 + 2];
  d[3] = g[seg * 4 + 3] * (v3 - mu) * inv + b[seg * 4 + 3];
  __syncthreads();
}

// mLSTM q/k/v/i/f matmuls: 8 output tiles, one per wave.
template <int KTX, int KTC>
__device__ void mlstm_matmuls(const _Float16* XN, int stride,
                              const _Float16* Wq, const _Float16* Wk,
                              const _Float16* Wv, const _Float16* Wi,
                              const _Float16* Wf,
                              const float* bq, const float* bk, const float* bv,
                              const float* bi, const float* bf, float mscale,
                              float* QT, float* KTb, float* VTb,
                              float* ITb, float* FTb) {
  int wave = threadIdx.x >> 5;
  if (wave < 4) {
    v8f acc = gemm<KTX>(XN, stride, Wq + wave * KTX * 512, vzero());
    store_tile(QT, 64, wave * 16, acc,
               [&](float v, int, int c) { return fast_tanh(v + bq[c]) * 0.5f; });
  } else if (wave == 4) {
    v8f acc = gemm<KTX>(XN, stride, Wk, vzero());
    store_tile(KTb, 16, 0, acc,
               [&](float v, int, int c) { return fast_tanh(v + bk[c]) * mscale; });
  } else if (wave == 5) {
    v8f acc = gemm<KTX>(XN, stride, Wv, vzero());
    store_tile(VTb, 16, 0, acc,
               [&](float v, int, int c) { return fast_tanh(v + bv[c]) * 0.5f; });
  } else if (wave == 6) {
    v8f acc = gemm<KTC>(XN, stride, Wi, vzero());
    store_tile(ITb, 16, 0, acc,
               [&](float v, int, int c) { return sig(v + bi[c]) * 0.9f + 0.05f; });
  } else {
    v8f acc = gemm<KTC>(XN, stride, Wf, vzero());
    store_tile(FTb, 16, 0, acc,
               [&](float v, int, int c) { return sig(v + bf[c]) * 0.9f + 0.05f; });
  }
}

// 16x16 memory-matrix update + m_mean. Thread t owns (row b=t/16, i=t%16, all j).
__device__ void mem_update(float (&m)[16], const float* KTb, const float* VTb,
                           const float* ITb, const float* FTb,
                           float* red, float* MM) {
  int b = threadIdx.x >> 4, r = threadIdx.x & 15;
  float fv = FTb[b * 16 + r], iv = ITb[b * 16 + r], vv = VTb[b * 16 + r];
  float s = 0.f;
#pragma unroll
  for (int j = 0; j < 16; ++j) {
    m[j] = fv * m[j] + iv * vv * KTb[b * 16 + j] * 0.1f;
    s += m[j];
  }
  red[b * 16 + r] = s;
  __syncthreads();
  if (threadIdx.x < 16) {
    float t = 0.f;
#pragma unroll
    for (int i = 0; i < 16; ++i) t += red[threadIdx.x * 16 + i];
    MM[threadIdx.x] = t * (1.0f / 256.0f);
  }
  __syncthreads();
}

// mLSTM output gate: xn@Wo[0:K] + m_mean * Wo[last] (rank-1 fixup in functor)
template <int KTX>
__device__ void mlstm_out(const _Float16* XN, int stride, const _Float16* Wo,
                          const float* bo, const float* wol, const float* MM,
                          float* OTb) {
  int wave = threadIdx.x >> 5;
  if (wave < 4) {
    v8f acc = gemm<KTX>(XN, stride, Wo + wave * KTX * 512, vzero());
    store_tile(OTb, 64, wave * 16, acc, [&](float v, int r, int c) {
      return sig(v + bo[c] + MM[r] * wol[c]) * 0.9f + 0.05f;
    });
  }
  __syncthreads();
}

// Full sLSTM step: 16 output tiles (4 gates x 4 col-tiles) over 8 waves.
__device__ void slstm_step(const _Float16* XN, const _Float16* W,
                           const float* Bs, const float* lng, const float* lnb,
                           float* C, float* H, float* ITg, float* FTg,
                           float* OTg, float* CHg, float* QT, float* red) {
  int wave = threadIdx.x >> 5;
  for (int tix = wave; tix < 16; tix += 8) {
    int mat = tix >> 2, ni = tix & 3;
    v8f acc = gemm<4>(XN, 128, W + mat * 8192 + ni * 2048, vzero());
    float* dst = (mat == 0) ? ITg : (mat == 1) ? FTg : (mat == 2) ? OTg : CHg;
    const float* bb = Bs + mat * 64;
    if (mat == 3)
      store_tile(dst, 64, ni * 16, acc,
                 [&](float v, int, int c) { return fast_tanh(v + bb[c]) * 0.5f; });
    else
      store_tile(dst, 64, ni * 16, acc, [&](float v, int, int c) {
        return sig(v + bb[c]) * 0.9f + 0.05f;
      });
  }
  __syncthreads();
  for (int e = threadIdx.x; e < 1024; e += 256)
    C[e] = FTg[e] * C[e] + ITg[e] * CHg[e];
  __syncthreads();
  layernorm_16x64(C, QT, lng, lnb, red);
  for (int e = threadIdx.x; e < 1024; e += 256)
    H[e] = OTg[e] * fast_tanh(QT[e]);
  __syncthreads();
}

// ---------------- kernel parameters ----------------
struct P {
  const float* x;
  const float *ip_b, *ip_w;
  const float *ln1_b, *ln1_g, *ln2_b, *ln2_g, *ln3_b, *ln3_g;
  const float *m1_f_b, *m1_f_w, *m1_i_b, *m1_i_w, *m1_k_b, *m1_k_w;
  const float *m1_o_b, *m1_o_w, *m1_q_b, *m1_q_w, *m1_v_b, *m1_v_w;
  const float *m1_ln_b, *m1_ln_g;
  const float *m2_f_b, *m2_f_w, *m2_i_b, *m2_i_w, *m2_k_b, *m2_k_w;
  const float *m2_o_b, *m2_o_w, *m2_q_b, *m2_q_w, *m2_v_b, *m2_v_w;
  const float *m2_ln_b, *m2_ln_g;
  const float *out_b, *out_w;
  const float *s1_c_b, *s1_c_w, *s1_f_b, *s1_f_w, *s1_i_b, *s1_i_w;
  const float *s1_o_b, *s1_o_w, *s1_ln_b, *s1_ln_g;
  const float *s2_c_b, *s2_c_w, *s2_f_b, *s2_f_w, *s2_i_b, *s2_i_w;
  const float *s2_o_b, *s2_o_w, *s2_ln_b, *s2_ln_g;
  float* out;
};

__global__ __launch_bounds__(256)
void xlstm_persistent_kernel(P p) {
  extern __shared__ __align__(16) char smem[];
  _Float16* hb = (_Float16*)smem;
  float* fb = (float*)(smem + (size_t)H_TOTAL * 2);

  const int tid = threadIdx.x;
  const int bb  = blockIdx.x * 16;  // batch-row base for this block

  // ---- one-time init: swizzle weights to f16 LDS fragments ----
  load_weight(p.m1_q_w, hb + HW_M1_Q, 128, 64);
  load_weight(p.m1_k_w, hb + HW_M1_K, 128, 16);
  load_weight(p.m1_v_w, hb + HW_M1_V, 128, 16);
  load_weight(p.m1_i_w, hb + HW_M1_I, 192, 16);
  load_weight(p.m1_f_w, hb + HW_M1_F, 192, 16);
  load_weight(p.m1_o_w, hb + HW_M1_O, 128, 64);  // rows 0..127 of 129
  load_weight(p.m2_q_w, hb + HW_M2_Q, 64, 64);
  load_weight(p.m2_k_w, hb + HW_M2_K, 64, 16);
  load_weight(p.m2_v_w, hb + HW_M2_V, 64, 16);
  load_weight(p.m2_i_w, hb + HW_M2_I, 128, 16);
  load_weight(p.m2_f_w, hb + HW_M2_F, 128, 16);
  load_weight(p.m2_o_w, hb + HW_M2_O, 64, 64);   // rows 0..63 of 65
  load_weight(p.s1_i_w, hb + HW_S1 + 0,     128, 64);
  load_weight(p.s1_f_w, hb + HW_S1 + 8192,  128, 64);
  load_weight(p.s1_o_w, hb + HW_S1 + 16384, 128, 64);
  load_weight(p.s1_c_w, hb + HW_S1 + 24576, 128, 64);
  load_weight(p.s2_i_w, hb + HW_S2 + 0,     128, 64);
  load_weight(p.s2_f_w, hb + HW_S2 + 8192,  128, 64);
  load_weight(p.s2_o_w, hb + HW_S2 + 16384, 128, 64);
  load_weight(p.s2_c_w, hb + HW_S2 + 24576, 128, 64);

  copyf(p.m1_q_b, fb + F_B_M1_Q, 64);
  copyf(p.m1_k_b, fb + F_B_M1_K, 16);
  copyf(p.m1_v_b, fb + F_B_M1_V, 16);
  copyf(p.m1_i_b, fb + F_B_M1_I, 16);
  copyf(p.m1_f_b, fb + F_B_M1_F, 16);
  copyf(p.m1_o_b, fb + F_B_M1_O, 64);
  copyf(p.m1_o_w + 128 * 64, fb + F_WO1L, 64);
  copyf(p.m1_ln_g, fb + F_LNM1_G, 64);
  copyf(p.m1_ln_b, fb + F_LNM1_B, 64);
  copyf(p.m2_q_b, fb + F_B_M2_Q, 64);
  copyf(p.m2_k_b, fb + F_B_M2_K, 16);
  copyf(p.m2_v_b, fb + F_B_M2_V, 16);
  copyf(p.m2_i_b, fb + F_B_M2_I, 16);
  copyf(p.m2_f_b, fb + F_B_M2_F, 16);
  copyf(p.m2_o_b, fb + F_B_M2_O, 64);
  copyf(p.m2_o_w + 64 * 64, fb + F_WO2L, 64);
  copyf(p.m2_ln_g, fb + F_LNM2_G, 64);
  copyf(p.m2_ln_b, fb + F_LNM2_B, 64);
  copyf(p.s1_i_b, fb + F_BS1 + 0,   64);
  copyf(p.s1_f_b, fb + F_BS1 + 64,  64);
  copyf(p.s1_o_b, fb + F_BS1 + 128, 64);
  copyf(p.s1_c_b, fb + F_BS1 + 192, 64);
  copyf(p.s1_ln_g, fb + F_LNS1_G, 64);
  copyf(p.s1_ln_b, fb + F_LNS1_B, 64);
  copyf(p.s2_i_b, fb + F_BS2 + 0,   64);
  copyf(p.s2_f_b, fb + F_BS2 + 64,  64);
  copyf(p.s2_o_b, fb + F_BS2 + 128, 64);
  copyf(p.s2_c_b, fb + F_BS2 + 192, 64);
  copyf(p.s2_ln_g, fb + F_LNS2_G, 64);
  copyf(p.s2_ln_b, fb + F_LNS2_B, 64);
  copyf(p.ln1_g, fb + F_LN1_G, 64);
  copyf(p.ln1_b, fb + F_LN1_B, 64);
  copyf(p.ln2_g, fb + F_LN2_G, 64);
  copyf(p.ln2_b, fb + F_LN2_B, 64);
  copyf(p.ln3_g, fb + F_LN3_G, 64);
  copyf(p.ln3_b, fb + F_LN3_B, 64);
  copyf(p.ip_w, fb + F_WIP, 128);
  copyf(p.ip_b, fb + F_BIP, 128);

  // zero recurrent state
  for (int e = tid; e < 1024; e += 256) {
    fb[F_H1 + e] = 0.f; fb[F_H2 + e] = 0.f;
    fb[F_H3 + e] = 0.f; fb[F_H4 + e] = 0.f;
    fb[F_C3 + e] = 0.f; fb[F_C4 + e] = 0.f;
  }
  float m1s[16], m2s[16];
#pragma unroll
  for (int j = 0; j < 16; ++j) { m1s[j] = 0.f; m2s[j] = 0.f; }
  __syncthreads();

  _Float16* XN1 = hb + HX_N1;  // [16][192]
  _Float16* XN2 = hb + HX_N2;  // [16][128]
  _Float16* XN3 = hb + HX_N3;  // [16][128]
  _Float16* XN4 = hb + HX_N4;  // [16][128]
  float* QT  = fb + F_QT;   float* OTb = fb + F_OT;
  float* KTb = fb + F_KT;   float* VTb = fb + F_VT;
  float* ITb = fb + F_IT;   float* FTb = fb + F_FT;
  float* ITg = fb + F_ITG;  float* FTg = fb + F_FTG;
  float* OTg = fb + F_OTG;  float* CHg = fb + F_CHG;
  float* H1 = fb + F_H1; float* H2 = fb + F_H2;
  float* H3 = fb + F_H3; float* H4 = fb + F_H4;
  float* C3 = fb + F_C3; float* C4 = fb + F_C4;
  float* RED = fb + F_RED; float* MM = fb + F_MM;

  // ---- recurrence over time ----
  for (int t = 0; t < Tlen; ++t) {
    // stage x_t for the 16 rows; prefetch next timestep
    if (tid < 16) {
      RED[tid] = p.x[(size_t)(bb + tid) * Tlen + t];
      if (t + 1 < Tlen)
        __builtin_prefetch(&p.x[(size_t)(bb + tid) * Tlen + t + 1], 0, 0);
    }
    __syncthreads();

    // input proj + m1 xn into comb1 (cols 0..127), tanh(h1)*0.5 (cols 128..191)
    for (int e = tid; e < 2048; e += 256) {
      int b = e >> 7, j = e & 127;
      float xp = fast_tanh(RED[b] * fb[F_WIP + j] + fb[F_BIP + j]) * 0.5f;
      XN1[b * 192 + j] = (_Float16)(fast_tanh(xp) * 0.3f);
    }
    for (int e = tid; e < 1024; e += 256) {
      int b = e >> 6, j = e & 63;
      XN1[b * 192 + 128 + j] = (_Float16)(fast_tanh(H1[e]) * 0.5f);
    }
    __syncthreads();

    // ---- mLSTM 1 (din=128, comb K=192) ----
    mlstm_matmuls<4, 6>(XN1, 192, hb + HW_M1_Q, hb + HW_M1_K, hb + HW_M1_V,
                        hb + HW_M1_I, hb + HW_M1_F,
                        fb + F_B_M1_Q, fb + F_B_M1_K, fb + F_B_M1_V,
                        fb + F_B_M1_I, fb + F_B_M1_F, 0.25f,
                        QT, KTb, VTb, ITb, FTb);
    __syncthreads();
    mem_update(m1s, KTb, VTb, ITb, FTb, RED, MM);
    mlstm_out<4>(XN1, 192, hb + HW_M1_O, fb + F_B_M1_O, fb + F_WO1L, MM, OTb);
    layernorm_16x64(QT, QT, fb + F_LNM1_G, fb + F_LNM1_B, RED);
    for (int e = tid; e < 1024; e += 256) H1[e] = OTb[e] * fast_tanh(QT[e]);
    __syncthreads();

    // ln1 -> comb2
    layernorm_16x64(H1, QT, fb + F_LN1_G, fb + F_LN1_B, RED);
    for (int e = tid; e < 1024; e += 256) {
      int b = e >> 6, j = e & 63;
      XN2[b * 128 + j]      = (_Float16)(fast_tanh(QT[e]) * 0.3f);
      XN2[b * 128 + 64 + j] = (_Float16)(fast_tanh(H2[e]) * 0.5f);
    }
    __syncthreads();

    // ---- mLSTM 2 (din=64, comb K=128) ----
    mlstm_matmuls<2, 4>(XN2, 128, hb + HW_M2_Q, hb + HW_M2_K, hb + HW_M2_V,
                        hb + HW_M2_I, hb + HW_M2_F,
                        fb + F_B_M2_Q, fb + F_B_M2_K, fb + F_B_M2_V,
                        fb + F_B_M2_I, fb + F_B_M2_F, 0.25f,
                        QT, KTb, VTb, ITb, FTb);
    __syncthreads();
    mem_update(m2s, KTb, VTb, ITb, FTb, RED, MM);
    mlstm_out<2>(XN2, 128, hb + HW_M2_O, fb + F_B_M2_O, fb + F_WO2L, MM, OTb);
    layernorm_16x64(QT, QT, fb + F_LNM2_G, fb + F_LNM2_B, RED);
    for (int e = tid; e < 1024; e += 256) H2[e] = OTb[e] * fast_tanh(QT[e]);
    __syncthreads();

    // ln2 -> comb3
    layernorm_16x64(H2, QT, fb + F_LN2_G, fb + F_LN2_B, RED);
    for (int e = tid; e < 1024; e += 256) {
      int b = e >> 6, j = e & 63;
      XN3[b * 128 + j]      = (_Float16)(fast_tanh(QT[e]) * 0.3f);
      XN3[b * 128 + 64 + j] = (_Float16)(fast_tanh(H3[e]) * 0.5f);
    }
    __syncthreads();

    // ---- sLSTM 1 ----
    slstm_step(XN3, hb + HW_S1, fb + F_BS1, fb + F_LNS1_G, fb + F_LNS1_B,
               C3, H3, ITg, FTg, OTg, CHg, QT, RED);

    // ln3 -> comb4
    layernorm_16x64(H3, QT, fb + F_LN3_G, fb + F_LN3_B, RED);
    for (int e = tid; e < 1024; e += 256) {
      int b = e >> 6, j = e & 63;
      XN4[b * 128 + j]      = (_Float16)(fast_tanh(QT[e]) * 0.3f);
      XN4[b * 128 + 64 + j] = (_Float16)(fast_tanh(H4[e]) * 0.5f);
    }
    __syncthreads();

    // ---- sLSTM 2 ----
    slstm_step(XN4, hb + HW_S2, fb + F_BS2, fb + F_LNS2_G, fb + F_LNS2_B,
               C4, H4, ITg, FTg, OTg, CHg, QT, RED);
  }

  // ---- final projection: out = h4 @ Wout + b ----
  if (tid < 16) {
    float s = p.out_b[0];
#pragma unroll 4
    for (int j = 0; j < 64; ++j) s += H4[tid * 64 + j] * p.out_w[j];
    p.out[bb + tid] = s;
  }
}

// ---------------- host entry ----------------
extern "C" void kernel_launch(void* const* d_in, const int* in_sizes, int n_in,
                              void* d_out, int out_size, void* d_ws,
                              size_t ws_size, hipStream_t stream) {
  (void)in_sizes; (void)n_in; (void)out_size; (void)d_ws; (void)ws_size;
  auto f = [&](int i) { return (const float*)d_in[i]; };
  P p;
  // Flattening: x first, then params with JAX-style sorted dict keys.
  p.x = f(0);
  p.ip_b = f(1);  p.ip_w = f(2);
  p.ln1_b = f(3); p.ln1_g = f(4);
  p.ln2_b = f(5); p.ln2_g = f(6);
  p.ln3_b = f(7); p.ln3_g = f(8);
  p.m1_f_b = f(9);  p.m1_f_w = f(10);
  p.m1_i_b = f(11); p.m1_i_w = f(12);
  p.m1_k_b = f(13); p.m1_k_w = f(14);
  p.m1_o_b = f(15); p.m1_o_w = f(16);
  p.m1_q_b = f(17); p.m1_q_w = f(18);
  p.m1_v_b = f(19); p.m1_v_w = f(20);
  p.m1_ln_b = f(21); p.m1_ln_g = f(22);
  p.m2_f_b = f(23); p.m2_f_w = f(24);
  p.m2_i_b = f(25); p.m2_i_w = f(26);
  p.m2_k_b = f(27); p.m2_k_w = f(28);
  p.m2_o_b = f(29); p.m2_o_w = f(30);
  p.m2_q_b = f(31); p.m2_q_w = f(32);
  p.m2_v_b = f(33); p.m2_v_w = f(34);
  p.m2_ln_b = f(35); p.m2_ln_g = f(36);
  p.out_b = f(37); p.out_w = f(38);
  p.s1_c_b = f(39); p.s1_c_w = f(40);
  p.s1_f_b = f(41); p.s1_f_w = f(42);
  p.s1_i_b = f(43); p.s1_i_w = f(44);
  p.s1_o_b = f(45); p.s1_o_w = f(46);
  p.s1_ln_b = f(47); p.s1_ln_g = f(48);
  p.s2_c_b = f(49); p.s2_c_w = f(50);
  p.s2_f_b = f(51); p.s2_f_w = f(52);
  p.s2_i_b = f(53); p.s2_i_w = f(54);
  p.s2_o_b = f(55); p.s2_o_w = f(56);
  p.s2_ln_b = f(57); p.s2_ln_g = f(58);
  p.out = (float*)d_out;

  dim3 grid(32), block(256);  // 512 rows / 16-row tiles, 8 wave32s per block
  xlstm_persistent_kernel<<<grid, block, SMEM_BYTES, stream>>>(p);
}
